// QbitNexus_15814069584150
// MI455X (gfx1250) — matrix operand order repaired and checked by reading, not compiled, not run on hardware
//
#include <hip/hip_runtime.h>
#include <hip/hip_bf16.h>
#include <math.h>

#define B_  2
#define S_  2048
#define H_  768
#define NV_ 12
#define NL_ 2
#define NH_ 8
#define HD_ 96
#define M_  (B_*S_)     // 4096
#define FF_ (4*H_)      // 3072

typedef __attribute__((ext_vector_type(16))) __bf16       v16bf;
typedef __attribute__((ext_vector_type(8)))  float        v8f;
typedef __attribute__((ext_vector_type(4)))  unsigned int u32x4;

union FragU { u32x4 q[2]; v16bf v; };

__device__ __forceinline__ unsigned short f2bf_bits(float f) {
  unsigned u = __builtin_bit_cast(unsigned, f);
  return (unsigned short)((u + 0x7FFFu + ((u >> 16) & 1u)) >> 16);
}

// ---- CDNA5 async global->LDS DMA (ASYNCcnt-tracked) ----
__device__ __forceinline__ void async_b128(const unsigned short* g, const unsigned short* l) {
  unsigned lo = (unsigned)(unsigned long long)l;           // LDS byte offset (low 32b of generic addr)
  unsigned long long ga = (unsigned long long)g;
  asm volatile("global_load_async_to_lds_b128 %0, %1, off" :: "v"(lo), "v"(ga) : "memory");
}
__device__ __forceinline__ void wait_async0() { asm volatile("s_wait_asynccnt 0" ::: "memory"); }
__device__ __forceinline__ void wait_async3() { asm volatile("s_wait_asynccnt 3" ::: "memory"); }

// =====================================================================
// WMMA GEMM:  C[M,N] = act(A[M,K](bf16) * Bt[N,K](bf16)^T + bias) (+resid)
// block tile 128x64, K-step 32, 8 waves (4x2), double-buffered async LDS
// =====================================================================
__global__ void __launch_bounds__(256)
k_gemm(const unsigned short* __restrict__ A, const unsigned short* __restrict__ Bt,
       const float* __restrict__ bias, const float* __restrict__ resid,
       float* __restrict__ Cf, unsigned short* __restrict__ Ch,
       int Mdim, int Ndim, int Kdim, int act)
{
  __shared__ __align__(16) unsigned short As[2][128][40];
  __shared__ __align__(16) unsigned short Bs[2][64][40];
  const int t  = threadIdx.x;
  const int m0 = blockIdx.y * 128;
  const int n0 = blockIdx.x * 64;
  const int nk = Kdim / 32;

  auto issue = [&](int buf, int k0) {
#pragma unroll
    for (int p = 0; p < 2; ++p) {                 // A: 128 rows x 64B = 512 chunks
      int idx = p * 256 + t;
      int row = idx >> 2, c = idx & 3;
      async_b128(A + (size_t)(m0 + row) * Kdim + k0 + c * 8, &As[buf][row][c * 8]);
    }
    { int row = t >> 2, c = t & 3;                // Bt: 64 rows x 64B = 256 chunks
      async_b128(Bt + (size_t)(n0 + row) * Kdim + k0 + c * 8, &Bs[buf][row][c * 8]); }
  };

  const int lane = t & 31, wid = t >> 5;
  const int hl = lane >> 4, l16 = lane & 15;
  const int wmb = (wid & 3) * 32, wnb = (wid >> 2) * 32;

  v8f acc[2][2];
  const v8f vzero = {0.f,0.f,0.f,0.f,0.f,0.f,0.f,0.f};
#pragma unroll
  for (int i = 0; i < 2; ++i)
#pragma unroll
    for (int j = 0; j < 2; ++j) acc[i][j] = vzero;

  issue(0, 0);
  for (int kt = 0; kt < nk; ++kt) {
    int buf = kt & 1;
    if (kt + 1 < nk) { issue(buf ^ 1, (kt + 1) * 32); wait_async3(); }
    else             { wait_async0(); }
    __syncthreads();

    FragU af[2], bf2[2];
#pragma unroll
    for (int i = 0; i < 2; ++i) {
      const unsigned short* p = &As[buf][wmb + i * 16 + l16][hl * 8];
      af[i].q[0] = *(const u32x4*)p;
      af[i].q[1] = *(const u32x4*)(p + 16);
    }
#pragma unroll
    for (int j = 0; j < 2; ++j) {
      const unsigned short* p = &Bs[buf][wnb + j * 16 + l16][hl * 8];
      bf2[j].q[0] = *(const u32x4*)p;
      bf2[j].q[1] = *(const u32x4*)(p + 16);
    }
#pragma unroll
    for (int i = 0; i < 2; ++i)
#pragma unroll
      for (int j = 0; j < 2; ++j)
        acc[i][j] = __builtin_amdgcn_wmma_f32_16x16x32_bf16(
            false, af[i].v, false, bf2[j].v, (short)0, acc[i][j], false, false);
    __syncthreads();
  }

#pragma unroll
  for (int i = 0; i < 2; ++i)
#pragma unroll
    for (int j = 0; j < 2; ++j)
#pragma unroll
      for (int r = 0; r < 8; ++r) {
        int row = m0 + wmb + i * 16 + hl * 8 + r;
        int col = n0 + wnb + j * 16 + l16;
        float v = acc[i][j][r] + bias[col];
        if (act) v = 0.5f * v * (1.0f + erff(v * 0.70710678118f));   // exact GELU
        if (resid) v += resid[(size_t)row * Ndim + col];
        if (Cf) Cf[(size_t)row * Ndim + col] = v;
        else    Ch[(size_t)row * Ndim + col] = f2bf_bits(v);
      }
  (void)Mdim;
}

// =====================================================================
// Flash attention: 4 waves/block, 16 q-rows/wave, 32-key tiles in LDS
// =====================================================================
__global__ void __launch_bounds__(128)
k_attn(const unsigned short* __restrict__ Q, const unsigned short* __restrict__ Kb,
       const unsigned short* __restrict__ Vb, const int* __restrict__ mask,
       float* __restrict__ O)
{
  __shared__ __align__(16) unsigned short Ks[32][104];  // [key][d], padded
  __shared__ __align__(16) unsigned short Vt[96][40];   // [d][key], padded
  __shared__ __align__(16) unsigned short Ps[4][16][40];
  const int t = threadIdx.x, lane = t & 31, wid = t >> 5;
  const int hl = lane >> 4, l16 = lane & 15;
  const int b = blockIdx.z, hh = blockIdx.y;
  const int q0 = blockIdx.x * 64 + wid * 16;

  FragU qf[3];
  {
    const size_t rowQ = (size_t)(b * S_ + q0 + l16) * H_ + hh * HD_;
#pragma unroll
    for (int g = 0; g < 3; ++g) {
      const unsigned short* p = Q + rowQ + g * 32 + hl * 8;
      qf[g].q[0] = *(const u32x4*)p;
      qf[g].q[1] = *(const u32x4*)(p + 16);
    }
  }

  v8f o[6];
  const v8f vzero = {0.f,0.f,0.f,0.f,0.f,0.f,0.f,0.f};
  float mrow[8], lrow[8];
#pragma unroll
  for (int j = 0; j < 6; ++j) o[j] = vzero;
#pragma unroll
  for (int r = 0; r < 8; ++r) { mrow[r] = -1e30f; lrow[r] = 0.f; }
  const float scale = rsqrtf((float)HD_);

  for (int kv = 0; kv < S_; kv += 32) {
#pragma unroll
    for (int p = 0; p < 3; ++p) {                      // K tile: 32x96 bf16, 384 chunks
      int idx = p * 128 + t;
      int row = idx / 12, c = idx % 12;
      async_b128(Kb + (size_t)(b * S_ + kv + row) * H_ + hh * HD_ + c * 8, &Ks[row][c * 8]);
    }
    for (int idx = t; idx < 32 * 96; idx += 128) {     // V tile transposed (scalar)
      int row = idx / 96, d = idx % 96;
      Vt[d][row] = Vb[(size_t)(b * S_ + kv + row) * H_ + hh * HD_ + d];
    }
    wait_async0();
    __syncthreads();

    v8f s2[2];
#pragma unroll
    for (int c2 = 0; c2 < 2; ++c2) {
      v8f sc = vzero;
#pragma unroll
      for (int g = 0; g < 3; ++g) {
        FragU kf;
        const unsigned short* p = &Ks[c2 * 16 + l16][g * 32 + hl * 8];
        kf.q[0] = *(const u32x4*)p;
        kf.q[1] = *(const u32x4*)(p + 16);
        sc = __builtin_amdgcn_wmma_f32_16x16x32_bf16(
            false, qf[g].v, false, kf.v, (short)0, sc, false, false);
      }
      int key = kv + c2 * 16 + l16;
      float kbias = mask[b * S_ + key] ? 0.0f : -1e30f;
#pragma unroll
      for (int r = 0; r < 8; ++r) sc[r] = sc[r] * scale + kbias;
      s2[c2] = sc;
    }

    float alpha[8];
#pragma unroll
    for (int r = 0; r < 8; ++r) {
      float mx = fmaxf(s2[0][r], s2[1][r]);
      for (int d = 1; d < 16; d <<= 1) mx = fmaxf(mx, __shfl_xor(mx, d, 32));
      float nm = fmaxf(mrow[r], mx);
      alpha[r] = __expf(mrow[r] - nm);
      float p0 = __expf(s2[0][r] - nm);
      float p1 = __expf(s2[1][r] - nm);
      s2[0][r] = p0; s2[1][r] = p1;
      float rs = p0 + p1;
      for (int d = 1; d < 16; d <<= 1) rs += __shfl_xor(rs, d, 32);
      lrow[r] = lrow[r] * alpha[r] + rs;
      mrow[r] = nm;
    }
#pragma unroll
    for (int j = 0; j < 6; ++j)
#pragma unroll
      for (int r = 0; r < 8; ++r) o[j][r] *= alpha[r];

    // P (C-layout) -> LDS -> A-layout fragment
#pragma unroll
    for (int c2 = 0; c2 < 2; ++c2)
#pragma unroll
      for (int r = 0; r < 8; ++r)
        Ps[wid][hl * 8 + r][c2 * 16 + l16] = f2bf_bits(s2[c2][r]);
    asm volatile("" ::: "memory");   // same-wave LDS is in-order
    FragU pf;
    { const unsigned short* p = &Ps[wid][l16][hl * 8];
      pf.q[0] = *(const u32x4*)p;
      pf.q[1] = *(const u32x4*)(p + 16); }
#pragma unroll
    for (int j = 0; j < 6; ++j) {
      FragU vf;
      const unsigned short* p = &Vt[j * 16 + l16][hl * 8];
      vf.q[0] = *(const u32x4*)p;
      vf.q[1] = *(const u32x4*)(p + 16);
      o[j] = __builtin_amdgcn_wmma_f32_16x16x32_bf16(
          false, pf.v, false, vf.v, (short)0, o[j], false, false);
    }
    __syncthreads();
  }

#pragma unroll
  for (int j = 0; j < 6; ++j)
#pragma unroll
    for (int r = 0; r < 8; ++r) {
      int row = q0 + hl * 8 + r;
      int col = hh * HD_ + j * 16 + l16;
      O[(size_t)(b * S_ + row) * H_ + col] = o[j][r] / lrow[r];
    }
}

// =====================================================================
// Small f32 kernels (preamble / glue)
// =====================================================================
__global__ void __launch_bounds__(128)
k_vsm(const float* __restrict__ x, const float* __restrict__ vaW,
      const float* __restrict__ vab, float* __restrict__ vw)
{
  int row = blockIdx.x;
  int t = threadIdx.x, lane = t & 31, wid = t >> 5;
  const float* xr = x + (size_t)row * H_;
  float acc[NV_];
#pragma unroll
  for (int v = 0; v < NV_; ++v) acc[v] = 0.f;
  for (int h0 = t; h0 < H_; h0 += 128) {
    float xv = xr[h0];
#pragma unroll
    for (int v = 0; v < NV_; ++v) acc[v] += xv * vaW[(size_t)h0 * NV_ + v];
  }
  __shared__ float red[4][NV_];
  __shared__ float sm[NV_];
#pragma unroll
  for (int v = 0; v < NV_; ++v) {
    float a = acc[v];
    for (int d = 1; d < 32; d <<= 1) a += __shfl_xor(a, d, 32);
    if (lane == 0) red[wid][v] = a;
  }
  __syncthreads();
  if (t == 0) {
    float s[NV_], mx = -1e30f;
    for (int v = 0; v < NV_; ++v) {
      s[v] = red[0][v] + red[1][v] + red[2][v] + red[3][v] + vab[v];
      mx = fmaxf(mx, s[v]);
    }
    float ss = 0.f;
    for (int v = 0; v < NV_; ++v) { s[v] = __expf(s[v] - mx); ss += s[v]; }
    for (int v = 0; v < NV_; ++v) sm[v] = s[v] / ss;
  }
  __syncthreads();
  if (t < NV_) vw[(size_t)row * NV_ + t] = sm[t];
}

__global__ void k_wmean(const float* __restrict__ x, const float* __restrict__ vw,
                        float* __restrict__ wm)
{
  int bv = blockIdx.x, b = bv / NV_, v = bv % NV_;
  int d = blockIdx.y * 256 + threadIdx.x;
  float acc = 0.f;
  for (int s = 0; s < S_; ++s)
    acc += x[(size_t)(b * S_ + s) * H_ + d] * vw[(size_t)(b * S_ + s) * NV_ + v];
  wm[(size_t)bv * H_ + d] = acc / (float)S_;
}

__global__ void __launch_bounds__(128)
k_qbit(const float* __restrict__ wm, const float* __restrict__ aW, const float* __restrict__ ab,
       const float* __restrict__ pW, const float* __restrict__ pb, const float* __restrict__ uW,
       float* __restrict__ amp, float* __restrict__ ph, float* __restrict__ st)
{
  int bv = blockIdx.x, v = bv % NV_;
  int d = blockIdx.y * 128 + threadIdx.x;
  __shared__ float w[H_];
  for (int h0 = threadIdx.x; h0 < H_; h0 += 128) w[h0] = wm[(size_t)bv * H_ + h0];
  __syncthreads();
  float a = 0.f, p = 0.f, u = 0.f;
  const float* aWv = aW + (size_t)v * H_ * H_;
  const float* pWv = pW + (size_t)v * H_ * H_;
  const float* uWv = uW + (size_t)v * H_ * H_;
  for (int h0 = 0; h0 < H_; ++h0) {
    float wv = w[h0];
    a += wv * aWv[(size_t)h0 * H_ + d];
    p += wv * pWv[(size_t)h0 * H_ + d];
    u += wv * uWv[(size_t)h0 * H_ + d];
  }
  size_t o = (size_t)bv * H_ + d;
  amp[o] = 1.0f / (1.0f + __expf(-(a + ab[v * H_ + d])));
  ph[o]  = tanhf(p + pb[v * H_ + d]) * 3.14159265358979f;
  st[o]  = u;
}

__global__ void k_state(const float* __restrict__ amp, const float* __restrict__ ph,
                        const float* __restrict__ st, float* __restrict__ state)
{
  int bv = blockIdx.x;
  int d = blockIdx.y * 256 + threadIdx.x;
  int dm = (d + H_ - 1) % H_;
  size_t o = (size_t)bv * H_;
  float c = st[o + d]  * amp[o + d]  * __cosf(ph[o + d]);
  float s = st[o + dm] * amp[o + dm] * __sinf(ph[o + dm]);
  state[o + d] = c + s;
}

__global__ void __launch_bounds__(256)
k_ln(const float* __restrict__ in, const float* __restrict__ g, const float* __restrict__ bb,
     float* __restrict__ out, int rows)
{
  int row = blockIdx.x; if (row >= rows) return;
  int t = threadIdx.x;
  const float* xr = in + (size_t)row * H_;
  __shared__ float red[256];
  float acc = 0.f;
  for (int d = t; d < H_; d += 256) acc += xr[d];
  red[t] = acc; __syncthreads();
  for (int s = 128; s > 0; s >>= 1) { if (t < s) red[t] += red[t + s]; __syncthreads(); }
  float mean = red[0] / (float)H_;
  __syncthreads();
  acc = 0.f;
  for (int d = t; d < H_; d += 256) { float v = xr[d] - mean; acc += v * v; }
  red[t] = acc; __syncthreads();
  for (int s = 128; s > 0; s >>= 1) { if (t < s) red[t] += red[t + s]; __syncthreads(); }
  float inv = rsqrtf(red[0] / (float)H_ + 1e-5f);
  for (int d = t; d < H_; d += 256)
    out[(size_t)row * H_ + d] = (xr[d] - mean) * inv * g[d] + bb[d];
}

__global__ void k_nexus(const float* __restrict__ vs, float* __restrict__ nexus)
{
  int b = blockIdx.x;
  int d = blockIdx.y * 256 + threadIdx.x;
  const float PHI = 1.6180339887498949f;
  float nrm = sqrtf(1.0f + PHI * PHI);
  float az[NV_] = {PHI/nrm,PHI/nrm,PHI/nrm,PHI/nrm, 0.f,0.f,0.f,0.f,
                   1.0f/nrm,1.0f/nrm,1.0f/nrm,1.0f/nrm};
  float gw[NV_], ssum = 0.f;
#pragma unroll
  for (int v = 0; v < NV_; ++v) { gw[v] = __powf(PHI, -az[v]); ssum += gw[v]; }
  float acc = 0.f;
#pragma unroll
  for (int v = 0; v < NV_; ++v)
    acc += vs[((size_t)b * NV_ + v) * H_ + d] * (gw[v] / ssum);
  nexus[(size_t)b * H_ + d] = acc;
}

__global__ void k_addb(const float* __restrict__ x, const float* __restrict__ nexus,
                       float* __restrict__ h)
{
  size_t idx = (size_t)blockIdx.x * 256 + threadIdx.x;
  int b = (int)(idx / ((size_t)S_ * H_));
  int d = (int)(idx % H_);
  h[idx] = x[idx] + nexus[b * H_ + d];
}

__global__ void k_cvt(const float* __restrict__ in, unsigned short* __restrict__ out, long n)
{
  long i = (long)blockIdx.x * 256 + threadIdx.x;
  if (i < n) out[i] = f2bf_bits(in[i]);
}

__global__ void k_cvtT(const float* __restrict__ in, unsigned short* __restrict__ out,
                       int K, int N)
{
  long i = (long)blockIdx.x * 256 + threadIdx.x;
  if (i >= (long)K * N) return;
  int nn = (int)(i / K), kk = (int)(i % K);
  out[i] = f2bf_bits(in[(size_t)kk * N + nn]);
}

__global__ void k_rmean(const float* __restrict__ h, float* __restrict__ hm)
{
  int b = blockIdx.x;
  int d = blockIdx.y * 256 + threadIdx.x;
  float acc = 0.f;
  for (int s = 0; s < S_; ++s) acc += h[(size_t)(b * S_ + s) * H_ + d];
  hm[b * H_ + d] = acc / (float)S_;
}

__global__ void __launch_bounds__(128)
k_nx(const float* __restrict__ hm, const float* __restrict__ nW, const float* __restrict__ nb,
     float* __restrict__ nx)
{
  int b = blockIdx.x;
  int d = blockIdx.y * 128 + threadIdx.x;
  __shared__ float w[H_];
  for (int h0 = threadIdx.x; h0 < H_; h0 += 128) w[h0] = hm[b * H_ + h0];
  __syncthreads();
  float acc = 0.f;
  for (int h0 = 0; h0 < H_; ++h0) acc += w[h0] * nW[(size_t)h0 * H_ + d];
  nx[b * H_ + d] = acc + nb[d];
}

__global__ void __launch_bounds__(256)
k_gate(float* __restrict__ ao, const float* __restrict__ nx,
       const float* __restrict__ gW, const float* __restrict__ gb)
{
  int row = blockIdx.x, b = row / S_;
  int t = threadIdx.x;
  __shared__ float red[256];
  float acc = 0.f;
  for (int d = t; d < H_; d += 256)
    acc += ao[(size_t)row * H_ + d] * gW[d] + nx[b * H_ + d] * gW[H_ + d];
  red[t] = acc; __syncthreads();
  for (int s = 128; s > 0; s >>= 1) { if (t < s) red[t] += red[t + s]; __syncthreads(); }
  float g = 1.0f / (1.0f + __expf(-(red[0] + gb[0])));
  for (int d = t; d < H_; d += 256) {
    size_t o = (size_t)row * H_ + d;
    ao[o] = ao[o] * (1.0f - g) + nx[b * H_ + d] * g;
  }
}

// =====================================================================
extern "C" void kernel_launch(void* const* d_in, const int* in_sizes, int n_in,
                              void* d_out, int out_size, void* d_ws, size_t ws_size,
                              hipStream_t stream)
{
  (void)in_sizes; (void)n_in; (void)out_size; (void)ws_size;
  const float* x    = (const float*)d_in[0];
  const int*   amsk = (const int*)  d_in[1];
  const float* qaW  = (const float*)d_in[2];
  const float* qab  = (const float*)d_in[3];
  const float* qpW  = (const float*)d_in[4];
  const float* qpb  = (const float*)d_in[5];
  const float* quW  = (const float*)d_in[6];
  const float* vaW  = (const float*)d_in[7];
  const float* vab  = (const float*)d_in[8];
  // d_in[9..11] (edge_coupling, face_W, face_b) and [14..17] feed dead code: skipped.
  const float* vng  = (const float*)d_in[12];
  const float* vnb  = (const float*)d_in[13];
  const float* ong  = (const float*)d_in[18];
  const float* onb  = (const float*)d_in[19];
  const float* qW   = (const float*)d_in[20];
  const float* qb   = (const float*)d_in[21];
  const float* kW   = (const float*)d_in[22];
  const float* kb   = (const float*)d_in[23];
  const float* vW   = (const float*)d_in[24];
  const float* vb   = (const float*)d_in[25];
  const float* oW   = (const float*)d_in[26];
  const float* ob   = (const float*)d_in[27];
  const float* nW   = (const float*)d_in[28];
  const float* nb   = (const float*)d_in[29];
  const float* gW   = (const float*)d_in[30];
  const float* gb   = (const float*)d_in[31];
  const float* W1   = (const float*)d_in[32];
  const float* b1   = (const float*)d_in[33];
  const float* W2   = (const float*)d_in[34];
  const float* b2   = (const float*)d_in[35];

  char* wp = (char*)d_ws;
  auto alloc = [&](size_t bytes) -> void* {
    void* p = wp; wp += (bytes + 255) & ~(size_t)255; return p;
  };
  float* vw    = (float*)alloc((size_t)M_ * NV_ * 4);
  float* wmb   = (float*)alloc((size_t)B_ * NV_ * H_ * 4);
  float* amp   = (float*)alloc((size_t)B_ * NV_ * H_ * 4);
  float* ph    = (float*)alloc((size_t)B_ * NV_ * H_ * 4);
  float* st    = (float*)alloc((size_t)B_ * NV_ * H_ * 4);
  float* state = (float*)alloc((size_t)B_ * NV_ * H_ * 4);
  float* vs    = (float*)alloc((size_t)B_ * NV_ * H_ * 4);
  float* nexus = (float*)alloc((size_t)B_ * H_ * 4);
  float* h     = (float*)alloc((size_t)M_ * H_ * 4);
  float* ao    = (float*)alloc((size_t)M_ * H_ * 4);
  float* hm    = (float*)alloc((size_t)B_ * H_ * 4);
  float* nx    = (float*)alloc((size_t)B_ * H_ * 4);
  unsigned short* hb   = (unsigned short*)alloc((size_t)M_ * H_ * 2);
  unsigned short* q16  = (unsigned short*)alloc((size_t)M_ * H_ * 2);
  unsigned short* k16  = (unsigned short*)alloc((size_t)M_ * H_ * 2);
  unsigned short* v16  = (unsigned short*)alloc((size_t)M_ * H_ * 2);
  unsigned short* ab16 = (unsigned short*)alloc((size_t)M_ * H_ * 2);
  unsigned short* h1b  = (unsigned short*)alloc((size_t)M_ * FF_ * 2);
  unsigned short* qWt  = (unsigned short*)alloc((size_t)NL_ * H_ * H_ * 2);
  unsigned short* kWt  = (unsigned short*)alloc((size_t)NL_ * H_ * H_ * 2);
  unsigned short* vWt  = (unsigned short*)alloc((size_t)NL_ * H_ * H_ * 2);
  unsigned short* oWt  = (unsigned short*)alloc((size_t)NL_ * H_ * H_ * 2);
  unsigned short* W1t  = (unsigned short*)alloc((size_t)H_ * FF_ * 2);
  unsigned short* W2t  = (unsigned short*)alloc((size_t)H_ * FF_ * 2);

  const long hh2 = (long)H_ * H_;
  const int  cvtHH = (int)((hh2 + 255) / 256);
  for (int l = 0; l < NL_; ++l) {
    k_cvtT<<<cvtHH, 256, 0, stream>>>(qW + l * hh2, qWt + l * hh2, H_, H_);
    k_cvtT<<<cvtHH, 256, 0, stream>>>(kW + l * hh2, kWt + l * hh2, H_, H_);
    k_cvtT<<<cvtHH, 256, 0, stream>>>(vW + l * hh2, vWt + l * hh2, H_, H_);
    k_cvtT<<<cvtHH, 256, 0, stream>>>(oW + l * hh2, oWt + l * hh2, H_, H_);
  }
  const long hf = (long)H_ * FF_;
  k_cvtT<<<(int)((hf + 255) / 256), 256, 0, stream>>>(W1, W1t, H_, FF_);
  k_cvtT<<<(int)((hf + 255) / 256), 256, 0, stream>>>(W2, W2t, FF_, H_);

  // ---- preamble ----
  k_vsm  <<<M_, 128, 0, stream>>>(x, vaW, vab, vw);
  k_wmean<<<dim3(B_ * NV_, H_ / 256), 256, 0, stream>>>(x, vw, wmb);
  k_qbit <<<dim3(B_ * NV_, H_ / 128), 128, 0, stream>>>(wmb, qaW, qab, qpW, qpb, quW, amp, ph, st);
  k_state<<<dim3(B_ * NV_, H_ / 256), 256, 0, stream>>>(amp, ph, st, state);
  k_ln   <<<B_ * NV_, 256, 0, stream>>>(state, vng, vnb, vs, B_ * NV_);
  k_nexus<<<dim3(B_, H_ / 256), 256, 0, stream>>>(vs, nexus);
  k_addb <<<(M_ * H_) / 256, 256, 0, stream>>>(x, nexus, h);

  const dim3 gH(H_ / 64, M_ / 128);
  const int cvtMH = (M_ * H_) / 256;

  // ---- transformer layers ----
  for (int l = 0; l < NL_; ++l) {
    k_cvt<<<cvtMH, 256, 0, stream>>>(h, hb, (long)M_ * H_);
    k_gemm<<<gH, 256, 0, stream>>>(hb, qWt + l * hh2, qb + l * H_, nullptr, nullptr, q16, M_, H_, H_, 0);
    k_gemm<<<gH, 256, 0, stream>>>(hb, kWt + l * hh2, kb + l * H_, nullptr, nullptr, k16, M_, H_, H_, 0);
    k_gemm<<<gH, 256, 0, stream>>>(hb, vWt + l * hh2, vb + l * H_, nullptr, nullptr, v16, M_, H_, H_, 0);
    k_rmean<<<dim3(B_, H_ / 256), 256, 0, stream>>>(h, hm);
    k_nx   <<<dim3(B_, H_ / 128), 128, 0, stream>>>(hm, nW + l * hh2, nb + l * H_, nx);
    k_attn <<<dim3(S_ / 64, NH_, B_), 128, 0, stream>>>(q16, k16, v16, amsk, ao);
    k_gate <<<M_, 256, 0, stream>>>(ao, nx, gW + (size_t)l * 2 * H_, gb + l);
    k_cvt  <<<cvtMH, 256, 0, stream>>>(ao, ab16, (long)M_ * H_);
    k_gemm <<<gH, 256, 0, stream>>>(ab16, oWt + l * hh2, ob + l * H_, h, h, nullptr, M_, H_, H_, 0);
  }

  // ---- FFN (GELU fused, bf16 intermediate) ----
  k_cvt <<<cvtMH, 256, 0, stream>>>(h, hb, (long)M_ * H_);
  k_gemm<<<dim3(FF_ / 64, M_ / 128), 256, 0, stream>>>(hb, W1t, b1, nullptr, nullptr, h1b, M_, FF_, H_, 1);
  k_gemm<<<gH, 256, 0, stream>>>(h1b, W2t, b2, h, h, nullptr, M_, H_, FF_, 0);

  // ---- final layernorm -> output ----
  k_ln<<<M_, 256, 0, stream>>>(h, ong, onb, (float*)d_out, M_);
}
